// Multi_Heads_Self_Attn_Q_KV_83021717832740
// MI455X (gfx1250) — matrix operand order repaired and checked by reading, not compiled, not run on hardware
//
#include <hip/hip_runtime.h>

typedef __attribute__((ext_vector_type(16))) __bf16 v16bf;
typedef __attribute__((ext_vector_type(8)))  __bf16 bf16x8;
typedef __attribute__((ext_vector_type(4)))  __bf16 bf16x4;
typedef __attribute__((ext_vector_type(8)))  float  v8f;
typedef __attribute__((ext_vector_type(4)))  unsigned int u32x4;
typedef __attribute__((ext_vector_type(8)))  int i32x8;
typedef __attribute__((ext_vector_type(4)))  int i32x4;

#define SPA   4096   // 16*16*16 spatial positions
#define C_CH  128
#define HDIM  256    // N_HEADS * D_K
#define NH    8
#define DK    32
#define EPS_IN 1e-5f

#if defined(__has_builtin)
#if __has_builtin(__builtin_amdgcn_tensor_load_to_lds)
#define HAVE_TDM 1
#if __has_include(<hip/amd_detail/amd_gfx1250_TDM.h>)
// therock-10.0 headers -> clang-23 6-arg form
#define TDM_LOAD(g0, g1, z4a, z4b) \
  __builtin_amdgcn_tensor_load_to_lds(g0, g1, z4a, z4b, (i32x8)(0), 0)
#else
// ROCm 7.2 clang-22 5-arg form
#define TDM_LOAD(g0, g1, z4a, z4b) \
  __builtin_amdgcn_tensor_load_to_lds(g0, g1, z4a, z4b, 0)
#endif
#endif
#endif

#if defined(__has_builtin) && __has_builtin(__builtin_amdgcn_s_wait_tensorcnt)
#define TDM_WAIT() __builtin_amdgcn_s_wait_tensorcnt(0)
#else
#define TDM_WAIT() asm volatile("s_wait_tensorcnt 0x0" ::: "memory")
#endif

static __device__ __forceinline__ v8f wmma_bf16(v16bf a, v16bf b, v8f c) {
  return __builtin_amdgcn_wmma_f32_16x16x32_bf16(false, a, false, b, (short)0, c, false, false);
}
static __device__ __forceinline__ bf16x8 ld8(const __bf16* p) {
  return *(const bf16x8*)p;
}
static __device__ __forceinline__ v16bf join8(bf16x8 lo, bf16x8 hi) {
  return __builtin_shufflevector(lo, hi, 0,1,2,3,4,5,6,7,8,9,10,11,12,13,14,15);
}
static __device__ __forceinline__ float sigmoid_fast(float x) {
  return __builtin_amdgcn_rcpf(1.f + __expf(-x));   // v_rcp_f32, no div_scale chain
}

// ---------------------------------------------------------------------------
// float -> bf16 elementwise (weights); n multiple of 4
// ---------------------------------------------------------------------------
__global__ __launch_bounds__(256) void f2bf_kernel(const float* __restrict__ in,
                                                   __bf16* __restrict__ out, int n4) {
  int i = blockIdx.x * 256 + threadIdx.x;
  if (i < n4) {
    float4 v = ((const float4*)in)[i];
    bf16x4 o; o[0] = (__bf16)v.x; o[1] = (__bf16)v.y; o[2] = (__bf16)v.z; o[3] = (__bf16)v.w;
    *(bf16x4*)(out + (size_t)i * 4) = o;
  }
}

// ---------------------------------------------------------------------------
// Instance norm over spatial dim for one channel per block; bf16 output
// ---------------------------------------------------------------------------
__global__ __launch_bounds__(256) void instnorm_kernel(const float* __restrict__ x,
                                                       const float* __restrict__ g,
                                                       const float* __restrict__ b,
                                                       __bf16* __restrict__ out) {
  const int c   = blockIdx.x;
  const int tid = threadIdx.x;
  const float* xc = x + (size_t)c * SPA;
  const float4* x4 = (const float4*)xc;
  float s = 0.f, ss = 0.f;
  #pragma unroll
  for (int i = tid; i < SPA / 4; i += 256) {
    float4 v = x4[i];
    s  += v.x + v.y + v.z + v.w;
    ss += v.x * v.x + v.y * v.y + v.z * v.z + v.w * v.w;
  }
  __shared__ float r0[256], r1[256];
  r0[tid] = s; r1[tid] = ss;
  __syncthreads();
  for (int off = 128; off > 0; off >>= 1) {
    if (tid < off) { r0[tid] += r0[tid + off]; r1[tid] += r1[tid + off]; }
    __syncthreads();
  }
  const float mu  = r0[0] * (1.f / SPA);
  const float var = r1[0] * (1.f / SPA) - mu * mu;   // biased variance
  const float inv = rsqrtf(var + EPS_IN);
  const float sc  = g[c] * inv;
  const float sh  = b[c] - mu * sc;
  __bf16* oc = out + (size_t)c * SPA;
  #pragma unroll
  for (int i = tid; i < SPA / 4; i += 256) {
    float4 v = x4[i];
    bf16x4 o;
    o[0] = (__bf16)(v.x * sc + sh); o[1] = (__bf16)(v.y * sc + sh);
    o[2] = (__bf16)(v.z * sc + sh); o[3] = (__bf16)(v.w * sc + sh);
    *(bf16x4*)(oc + (size_t)i * 4) = o;
  }
}

// ---------------------------------------------------------------------------
// WMMA GEMM: out[M,S] = A[M,K] * X[K,S].
//  - Weight slab A[m0..m0+63][0..K) staged to LDS via Tensor Data Mover
//    (one tensor_load_to_lds per block; cooperative-load fallback).
//  - X tile (32 x 128) staged to LDS transposed each K step.
//  - 4 m-tiles per block; compile-time epilogue flags.
// ---------------------------------------------------------------------------
template<bool BIAS, bool RES, bool RELU, bool OUTF, bool OUTB, bool TRANSB>
__global__ __launch_bounds__(256) void gemm_wmma_kernel(
    const __bf16* __restrict__ A, const __bf16* __restrict__ X,
    const float* __restrict__ bias, const float* __restrict__ resid,
    float* __restrict__ outF, __bf16* __restrict__ outB,
    int K, int Mtot) {
  const int lane = threadIdx.x & 31;
  const int wave = threadIdx.x >> 5;
  const int lr   = lane & 15;
  const int hi   = lane >> 4;
  const int nblk = blockIdx.x * 128;
  const int m0   = blockIdx.y * 64;

  __shared__ __bf16 ldsa[64 * 256];  // weight slab [64][K], row-major
  __shared__ __bf16 ldsx[128][40];   // X tile transposed: [col][k], padded stride

#if defined(HAVE_TDM)
  if (threadIdx.x < 32) {            // one TDM issue per block (wave 0)
    const unsigned lds_off = (unsigned)(size_t)(void*)ldsa;
    const unsigned long long ga =
        (unsigned long long)(size_t)(A + (size_t)m0 * K);
    u32x4 g0;
    g0[0] = 1u;                                            // count=1, user mode
    g0[1] = lds_off;                                       // lds_addr
    g0[2] = (unsigned)(ga & 0xFFFFFFFFu);                  // global_addr[31:0]
    g0[3] = (unsigned)((ga >> 32) & 0x1FFFFFFu) | (2u << 30); // addr[56:32] | type=2
    i32x8 g1;
    g1[0] = 1 << 16;                                       // data_size = 2 bytes
    g1[1] = (K & 0xFFFF) << 16;                            // tensor_dim0[15:0]
    g1[2] = 64 << 16;                                      // dim0 hi | tensor_dim1=64
    g1[3] = K << 16;                                       // dim1 hi | tile_dim0=K
    g1[4] = 64;                                            // tile_dim1=64, tile_dim2=0
    g1[5] = K;                                             // tensor_dim0_stride[31:0]
    g1[6] = 0;
    g1[7] = 0;
    i32x4 z4 = {0, 0, 0, 0};
    TDM_LOAD(g0, g1, z4, z4);
    TDM_WAIT();
  }
#else
  // cooperative fallback: coalesced b128 copy of the weight slab
  for (int cid = threadIdx.x; cid < 64 * K / 8; cid += 256) {
    const int row = cid / (K / 8);
    const int c8  = cid % (K / 8);
    *(bf16x8*)(ldsa + (size_t)row * K + c8 * 8) =
        ld8(A + (size_t)(m0 + row) * K + c8 * 8);
  }
#endif
  __syncthreads();

  v8f acc[4] = {{}, {}, {}, {}};
  for (int k0 = 0; k0 < K; k0 += 32) {
    // cooperative coalesced load of X[k0..k0+31][nblk..nblk+127] -> LDS^T
    #pragma unroll
    for (int c = 0; c < 2; ++c) {
      const int cid  = threadIdx.x * 2 + c;    // 512 chunks of 8 bf16
      const int krow = cid >> 4;
      const int c8   = cid & 15;
      const __bf16* src = X + (size_t)(k0 + krow) * SPA + nblk + c8 * 8;
      if (k0 + 32 < K) __builtin_prefetch(src + (size_t)32 * SPA, 0, 1);
      bf16x8 ch = ld8(src);
      #pragma unroll
      for (int e = 0; e < 8; ++e) ldsx[c8 * 8 + e][krow] = ch[e];
    }
    __syncthreads();

    const __bf16* bp = &ldsx[wave * 16 + lr][16 * hi];
    const v16bf bv = join8(ld8(bp), ld8(bp + 8));
    #pragma unroll
    for (int mt = 0; mt < 4; ++mt) {
      const __bf16* ap = ldsa + (size_t)(mt * 16 + lr) * K + k0 + 16 * hi;
      const v16bf av = join8(ld8(ap), ld8(ap + 8));
      acc[mt] = wmma_bf16(av, bv, acc[mt]);
    }
    __syncthreads();
  }

  const int n = nblk + wave * 16 + lr;
  #pragma unroll
  for (int mt = 0; mt < 4; ++mt) {
    #pragma unroll
    for (int j = 0; j < 8; ++j) {
      const int row = m0 + mt * 16 + j + 8 * hi;   // C/D layout
      float v = acc[mt][j];
      if (BIAS) v += bias[row];
      if (RES)  v += resid[(size_t)row * SPA + n];
      if (RELU) v = fmaxf(v, 0.f);
      if (OUTF) outF[(size_t)row * SPA + n] = v;
      if (OUTB) {
        if (TRANSB) outB[(size_t)n * Mtot + row] = (__bf16)v;
        else        outB[(size_t)row * SPA + n]  = (__bf16)v;
      }
    }
  }
}

// ---------------------------------------------------------------------------
// Sigmoid attention, streaming over keys. One wave per (head, 16-query tile).
// qT,kT stored [SPA, HDIM] bf16; v,o stored [HDIM, SPA] bf16.
// Per-wave LDS transpose buffer -> wave-local s_wait_dscnt, no block barrier.
// ---------------------------------------------------------------------------
__global__ __launch_bounds__(256) void attn_kernel(
    const __bf16* __restrict__ qT, const __bf16* __restrict__ kT,
    const __bf16* __restrict__ v, const int* __restrict__ mask,
    __bf16* __restrict__ o) {
  const int lane = threadIdx.x & 31;
  const int wave = threadIdx.x >> 5;
  const int lr   = lane & 15;
  const int hi   = lane >> 4;
  const int h    = blockIdx.y;
  const int s0   = (blockIdx.x * 8 + wave) * 16;

  const __bf16* vp = v + (size_t)h * DK * SPA;
  __bf16*       op = o + (size_t)h * DK * SPA;

  __shared__ __bf16 ldsw[8][16][32];   // per-wave sigmoid-weight tile

  const __bf16* qrow = qT + (size_t)(s0 + lr) * HDIM + h * DK;
  const v16bf aq = join8(ld8(qrow + 16 * hi), ld8(qrow + 16 * hi + 8));

  v8f o0 = {}, o1 = {};                // o accumulators for d=0..15 / d=16..31
  for (int key0 = 0; key0 < SPA; key0 += 32) {
    const __bf16* kr0 = kT + (size_t)(key0 + lr) * HDIM + h * DK + 16 * hi;
    const __bf16* kr1 = kr0 + (size_t)16 * HDIM;
    const v16bf bk0 = join8(ld8(kr0), ld8(kr0 + 8));
    const v16bf bk1 = join8(ld8(kr1), ld8(kr1 + 8));
    v8f c0 = {}, c1 = {};
    c0 = wmma_bf16(aq, bk0, c0);
    c1 = wmma_bf16(aq, bk1, c1);

    // sigmoid * mask, spill 16x32 tile to LDS (D-layout -> A-layout)
    const float m0v = (float)mask[key0 + lr];
    const float m1v = (float)mask[key0 + 16 + lr];
    #pragma unroll
    for (int j = 0; j < 8; ++j) {
      const int row = j + 8 * hi;
      ldsw[wave][row][lr]      = (__bf16)(m0v * sigmoid_fast(c0[j]));
      ldsw[wave][row][16 + lr] = (__bf16)(m1v * sigmoid_fast(c1[j]));
    }
    // wave-local LDS RAW fence (per-wave buffer; DS ops are in-order per wave)
    asm volatile("s_wait_dscnt 0x0" ::: "memory");

    const __bf16* wrow = &ldsw[wave][lr][0];
    const v16bf aw = join8(ld8(wrow + 16 * hi), ld8(wrow + 16 * hi + 8));
    const __bf16* vr0 = vp + (size_t)lr * SPA + key0 + 16 * hi;
    const __bf16* vr1 = vp + (size_t)(16 + lr) * SPA + key0 + 16 * hi;
    if (key0 + 32 < SPA) {
      __builtin_prefetch(vr0 + 32, 0, 1);
      __builtin_prefetch(vr1 + 32, 0, 1);
    }
    const v16bf bv0 = join8(ld8(vr0), ld8(vr0 + 8));
    const v16bf bv1 = join8(ld8(vr1), ld8(vr1 + 8));
    o0 = wmma_bf16(aw, bv0, o0);
    o1 = wmma_bf16(aw, bv1, o1);
  }

  #pragma unroll
  for (int j = 0; j < 8; ++j) {
    const int qq = s0 + j + 8 * hi;
    op[(size_t)lr * SPA + qq]        = (__bf16)o0[j];
    op[(size_t)(16 + lr) * SPA + qq] = (__bf16)o1[j];
  }
}

// ---------------------------------------------------------------------------
extern "C" void kernel_launch(void* const* d_in, const int* in_sizes, int n_in,
                              void* d_out, int out_size, void* d_ws, size_t ws_size,
                              hipStream_t stream) {
  (void)in_sizes; (void)n_in; (void)out_size; (void)ws_size;
  const float* x_q  = (const float*)d_in[0];
  const float* x_kv = (const float*)d_in[1];
  const int*   mask = (const int*)d_in[2];
  const float* qW   = (const float*)d_in[3];
  const float* kW   = (const float*)d_in[4];
  const float* vW   = (const float*)d_in[5];
  const float* oW   = (const float*)d_in[6];
  const float* gq   = (const float*)d_in[7];
  const float* bq   = (const float*)d_in[8];
  const float* gk   = (const float*)d_in[9];
  const float* bk   = (const float*)d_in[10];
  const float* gp   = (const float*)d_in[11];
  const float* bp   = (const float*)d_in[12];
  const float* W1   = (const float*)d_in[13];
  const float* b1   = (const float*)d_in[14];
  const float* W2   = (const float*)d_in[15];
  const float* b2   = (const float*)d_in[16];
  float* out = (float*)d_out;

  char* ws = (char*)d_ws;
  auto alloc = [&](size_t bytes) -> char* {
    char* p = ws; ws += (bytes + 255) & ~(size_t)255; return p;
  };
  __bf16* xqn  = (__bf16*)alloc((size_t)C_CH * SPA * 2);
  __bf16* xkvn = (__bf16*)alloc((size_t)C_CH * SPA * 2);
  __bf16* qWb  = (__bf16*)alloc((size_t)HDIM * C_CH * 2);
  __bf16* kWb  = (__bf16*)alloc((size_t)HDIM * C_CH * 2);
  __bf16* vWb  = (__bf16*)alloc((size_t)HDIM * C_CH * 2);
  __bf16* oWb  = (__bf16*)alloc((size_t)C_CH * HDIM * 2);
  __bf16* W1b  = (__bf16*)alloc((size_t)2 * C_CH * C_CH * 2);
  __bf16* W2b  = (__bf16*)alloc((size_t)C_CH * 2 * C_CH * 2);
  __bf16* qTb  = (__bf16*)alloc((size_t)SPA * HDIM * 2);   // [S, HDIM]
  __bf16* kTb  = (__bf16*)alloc((size_t)SPA * HDIM * 2);   // [S, HDIM]
  __bf16* vb   = (__bf16*)alloc((size_t)HDIM * SPA * 2);   // [HDIM, S]
  __bf16* ob   = (__bf16*)alloc((size_t)HDIM * SPA * 2);   // [HDIM, S]
  float*  attnf= (float*) alloc((size_t)C_CH * SPA * 4);
  __bf16* hnb  = (__bf16*)alloc((size_t)C_CH * SPA * 2);
  __bf16* h1b  = (__bf16*)alloc((size_t)2 * C_CH * SPA * 2);

  const dim3 blk(256);
  const int nw4 = HDIM * C_CH / 4;   // all weight matrices have 32768 elements
  f2bf_kernel<<<dim3((nw4 + 255) / 256), blk, 0, stream>>>(qW, qWb, nw4);
  f2bf_kernel<<<dim3((nw4 + 255) / 256), blk, 0, stream>>>(kW, kWb, nw4);
  f2bf_kernel<<<dim3((nw4 + 255) / 256), blk, 0, stream>>>(vW, vWb, nw4);
  f2bf_kernel<<<dim3((nw4 + 255) / 256), blk, 0, stream>>>(oW, oWb, nw4);
  f2bf_kernel<<<dim3((nw4 + 255) / 256), blk, 0, stream>>>(W1, W1b, nw4);
  f2bf_kernel<<<dim3((nw4 + 255) / 256), blk, 0, stream>>>(W2, W2b, nw4);

  instnorm_kernel<<<dim3(C_CH), blk, 0, stream>>>(x_q,  gq, bq, xqn);
  instnorm_kernel<<<dim3(C_CH), blk, 0, stream>>>(x_kv, gk, bk, xkvn);

  const dim3 g256(SPA / 128, HDIM / 64);      // M = 256
  const dim3 g128(SPA / 128, C_CH / 64);      // M = 128
  // projections: q,k transposed bf16 [S,HDIM]; v as [HDIM,S]
  gemm_wmma_kernel<false,false,false,false,true,true ><<<g256, blk, 0, stream>>>(
      qWb, xqn,  nullptr, nullptr, nullptr, qTb, C_CH, HDIM);
  gemm_wmma_kernel<false,false,false,false,true,true ><<<g256, blk, 0, stream>>>(
      kWb, xkvn, nullptr, nullptr, nullptr, kTb, C_CH, HDIM);
  gemm_wmma_kernel<false,false,false,false,true,false><<<g256, blk, 0, stream>>>(
      vWb, xkvn, nullptr, nullptr, nullptr, vb,  C_CH, HDIM);

  // streaming sigmoid attention
  attn_kernel<<<dim3(SPA / 128, NH), blk, 0, stream>>>(qTb, kTb, vb, mask, ob);

  // output projection + residual(x_q) -> attn (f32)
  gemm_wmma_kernel<false,true,false,true,false,false><<<g128, blk, 0, stream>>>(
      oWb, ob, nullptr, x_q, attnf, nullptr, HDIM, C_CH);

  // post-norm + MLP
  instnorm_kernel<<<dim3(C_CH), blk, 0, stream>>>(attnf, gp, bp, hnb);
  gemm_wmma_kernel<true,false,true,false,true,false><<<g256, blk, 0, stream>>>(
      W1b, hnb, b1, nullptr, nullptr, h1b, C_CH, HDIM);
  gemm_wmma_kernel<true,true,false,true,false,false><<<g128, blk, 0, stream>>>(
      W2b, h1b, b2, attnf, out, nullptr, 2 * C_CH, C_CH);
}
